// TreeLSTM_37864431681917
// MI455X (gfx1250) — matrix operand order repaired and checked by reading, not compiled, not run on hardware
//
#include <hip/hip_runtime.h>
#include <hip/hip_bf16.h>

#define NODES 255
#define EPSF 1e-5f
#define TILE 12288          // 16x768 bf16 elements
#define TILE2 24576         // 32x768 bf16 elements

typedef __attribute__((ext_vector_type(16))) __bf16 v16bf;
typedef __attribute__((ext_vector_type(8)))  float  v8f;
typedef __attribute__((ext_vector_type(8)))  unsigned int v8u;

__device__ __forceinline__ unsigned short f2bf(float f) {
  unsigned u = __float_as_uint(f);
  u += 0x7fffu + ((u >> 16) & 1u);           // round-to-nearest-even
  return (unsigned short)(u >> 16);
}
__device__ __forceinline__ unsigned pack2(float lo, float hi) {
  return (unsigned)f2bf(lo) | ((unsigned)f2bf(hi) << 16);
}
__device__ __forceinline__ float bf2f(unsigned short s) {
  return __uint_as_float(((unsigned)s) << 16);
}
__device__ __forceinline__ unsigned addpk(unsigned a, unsigned b) {   // packed bf16 add
  const float lo = bf2f((unsigned short)a) + bf2f((unsigned short)b);
  const float hi = bf2f((unsigned short)(a >> 16)) + bf2f((unsigned short)(b >> 16));
  return pack2(lo, hi);
}
__device__ __forceinline__ float sigm(float x) { return 1.0f / (1.0f + __expf(-x)); }

__device__ __forceinline__ v16bf mk_frag(uint4 a, uint4 b) {
  v8u u; u[0]=a.x; u[1]=a.y; u[2]=a.z; u[3]=a.w; u[4]=b.x; u[5]=b.y; u[6]=b.z; u[7]=b.w;
  return __builtin_bit_cast(v16bf, u);
}

// A fragment (16x32 bf16) from an LDS tile (rows x 768 bf16, row-major).
// lane<16 -> row=lane, K {0..7},{16..23}; lane>=16 -> K {8..15},{24..31}.
__device__ __forceinline__ v16bf a_frag_lds(const unsigned short* tile, int k0, int lane) {
  const unsigned short* p = tile + (lane & 15) * 768 + k0 + ((lane >> 4) << 3);
  return mk_frag(*(const uint4*)p, *(const uint4*)(p + 16));
}

// A fragment from a global bf16 row-major matrix (lda = 768 bf16).
__device__ __forceinline__ v16bf a_frag_gbf(const unsigned short* __restrict__ A,
                                            int row0, int k0, int lane) {
  const unsigned short* p = A + (size_t)(row0 + (lane & 15)) * 768 + k0 + ((lane >> 4) << 3);
  return mk_frag(*(const uint4*)p, *(const uint4*)(p + 16));
}

// B fragment (32x16): output col n = row (g0+n) of bf16 weight W (G x 768, row-major).
// lane<16 -> K {0..7},{16..23}; lane>=16 -> K {8..15},{24..31} of their column.
__device__ __forceinline__ v16bf b_frag(const unsigned short* __restrict__ W,
                                        int g0, int k0, int lane) {
  const unsigned short* p = W + (size_t)(g0 + (lane & 15)) * 768 + k0 + ((lane >> 4) << 3);
  return mk_frag(*(const uint4*)p, *(const uint4*)(p + 16));
}

#define WMMA_BF16(A, Bf, C) \
  __builtin_amdgcn_wmma_f32_16x16x32_bf16(false, (A), false, (Bf), (short)0, (C), false, false)

// i/o/u GEMMs over two 16-row subtiles sharing A-loads and B-fragments:
// per K-step: 2 LDS A-loads + 3 global B-loads -> 6 WMMAs.
__device__ __forceinline__ void mma3x2(v8f& i0, v8f& o0, v8f& u0,
                                       v8f& i1, v8f& o1, v8f& u1,
                                       const unsigned short* aT,
                                       const unsigned short* __restrict__ W, int g, int lane) {
  #pragma unroll 4
  for (int k0 = 0; k0 < 768; k0 += 32) {
    const v16bf a0 = a_frag_lds(aT,        k0, lane);
    const v16bf a1 = a_frag_lds(aT + TILE, k0, lane);
    const v16bf b0 = b_frag(W,                       g, k0, lane);
    const v16bf b1 = b_frag(W + (size_t)768  * 768,  g, k0, lane);
    const v16bf b2 = b_frag(W + (size_t)1536 * 768,  g, k0, lane);
    i0 = WMMA_BF16(a0, b0, i0); i1 = WMMA_BF16(a1, b0, i1);
    o0 = WMMA_BF16(a0, b1, o0); o1 = WMMA_BF16(a1, b1, o1);
    u0 = WMMA_BF16(a0, b2, u0); u1 = WMMA_BF16(a1, b2, u1);
  }
}

// fx GEMM over two subtiles: 2 A-loads + 1 B-load -> 2 WMMAs per K-step.
__device__ __forceinline__ void mma1x2(v8f& f0, v8f& f1, const unsigned short* aT,
                                       const unsigned short* __restrict__ W, int g, int lane) {
  #pragma unroll 4
  for (int k0 = 0; k0 < 768; k0 += 32) {
    const v16bf bf = b_frag(W, g, k0, lane);
    f0 = WMMA_BF16(a_frag_lds(aT,        k0, lane), bf, f0);
    f1 = WMMA_BF16(a_frag_lds(aT + TILE, k0, lane), bf, f1);
  }
}

// fh GEMMs: both children x both subtiles share one Wfh fragment:
// 4 A-loads + 1 B-load -> 4 WMMAs per K-step.
__device__ __forceinline__ void mma_fh(v8f& p00, v8f& p01, v8f& p10, v8f& p11,
                                       const unsigned short* c0T, const unsigned short* c1T,
                                       const unsigned short* __restrict__ W, int g, int lane) {
  #pragma unroll 4
  for (int k0 = 0; k0 < 768; k0 += 32) {
    const v16bf bf = b_frag(W, g, k0, lane);
    p00 = WMMA_BF16(a_frag_lds(c0T,        k0, lane), bf, p00);
    p01 = WMMA_BF16(a_frag_lds(c0T + TILE, k0, lane), bf, p01);
    p10 = WMMA_BF16(a_frag_lds(c1T,        k0, lane), bf, p10);
    p11 = WMMA_BF16(a_frag_lds(c1T + TILE, k0, lane), bf, p11);
  }
}

// fp32 -> bf16 bulk convert (8 elements per thread).
__global__ void __launch_bounds__(256)
cvt_bf16(const float* __restrict__ src, unsigned short* __restrict__ dst, long n8) {
  const long i = (long)blockIdx.x * 256 + threadIdx.x;
  if (i >= n8) return;
  const float4 a = ((const float4*)src)[2 * i], b = ((const float4*)src)[2 * i + 1];
  uint4 o;
  o.x = pack2(a.x, a.y); o.y = pack2(a.z, a.w);
  o.z = pack2(b.x, b.y); o.w = pack2(b.z, b.w);
  ((uint4*)dst)[i] = o;
}

// One tree level, fully fused: WMMA GEMMs + gates + c + LN + h + LN.
// Block = 32 rows (row = b*n + node) x all 768 hidden columns; 8 waves.
__global__ void __launch_bounds__(256)
treelstm_level(const unsigned short* __restrict__ xbf,
               const unsigned short* __restrict__ Wioux, const float* __restrict__ bioux,
               const unsigned short* __restrict__ Wiouh,
               const unsigned short* __restrict__ Wfx,   const float* __restrict__ bfx,
               const unsigned short* __restrict__ Wfh,   const float* __restrict__ bfh,
               const float* __restrict__ lncg,  const float* __restrict__ lncb,
               const float* __restrict__ lnhg,  const float* __restrict__ lnhb,
               const unsigned short* __restrict__ hprev, unsigned short* __restrict__ hout,
               int nlog2, int has_children) {
  extern __shared__ unsigned short smem[];
  unsigned short* xsT = smem;                // 32x768 bf16 : x rows
  unsigned short* hsT = smem + 1 * TILE2;    // h0+h1 (child sum)
  unsigned short* c0T = smem + 2 * TILE2;    // child 0 h
  unsigned short* c1T = smem + 3 * TILE2;    // child 1 h
  unsigned short* cT  = smem + 4 * TILE2;    // cell state (then h-raw)
  unsigned short* oT  = smem + 5 * TILE2;    // sigmoid(o)

  const int n   = 1 << nlog2;
  const int tid = threadIdx.x;
  const int m   = blockIdx.x;

  // ---- stage A operands into LDS (bf16, straight copies) ----
  for (int e = tid; e < 32 * 96; e += 256) {           // uint4 = 8 bf16 chunks
    const int r = e / 96, q = (e % 96) << 3;           // q: bf16 column
    const int rowg = m * 32 + r;
    const int b = rowg >> nlog2, node = rowg & (n - 1);
    *(uint4*)(xsT + r * 768 + q) =
        *(const uint4*)(xbf + ((size_t)b * NODES + (n - 1) + node) * 768 + q);
    if (has_children) {
      const unsigned short* hp = hprev + ((size_t)((b << (nlog2 + 1)) + 2 * node)) * 768 + q;
      const uint4 h0 = *(const uint4*)hp;
      const uint4 h1 = *(const uint4*)(hp + 768);
      *(uint4*)(c0T + r * 768 + q) = h0;
      *(uint4*)(c1T + r * 768 + q) = h1;
      uint4 hs;
      hs.x = addpk(h0.x, h1.x); hs.y = addpk(h0.y, h1.y);
      hs.z = addpk(h0.z, h1.z); hs.w = addpk(h0.w, h1.w);
      *(uint4*)(hsT + r * 768 + q) = hs;
    }
  }
  __syncthreads();

  // ---- per-wave 16-column slices over two 16-row subtiles ----
  const int wave = tid >> 5, lane = tid & 31;
  for (int j = wave; j < 48; j += 8) {
    const int g   = j << 4;
    const int col = g + (lane & 15);
    v8f i0 = {}, o0 = {}, u0 = {}, i1 = {}, o1 = {}, u1 = {};
    mma3x2(i0, o0, u0, i1, o1, u1, xsT, Wioux, g, lane);
    if (has_children) mma3x2(i0, o0, u0, i1, o1, u1, hsT, Wiouh, g, lane);
    const float bi = bioux[col], bog = bioux[768 + col], bu = bioux[1536 + col];
    v8f cf0, cf1, of0, of1;
    #pragma unroll
    for (int t = 0; t < 8; ++t) {
      cf0[t] = sigm(i0[t] + bi) * tanhf(u0[t] + bu);
      of0[t] = sigm(o0[t] + bog);
      cf1[t] = sigm(i1[t] + bi) * tanhf(u1[t] + bu);
      of1[t] = sigm(o1[t] + bog);
    }
    if (has_children) {
      v8f fx0 = {}, fx1 = {};
      mma1x2(fx0, fx1, xsT, Wfx, g, lane);
      v8f p00 = {}, p01 = {}, p10 = {}, p11 = {};
      mma_fh(p00, p01, p10, p11, c0T, c1T, Wfh, g, lane);
      const float bb = bfx[col] + bfh[col];
      #pragma unroll
      for (int t = 0; t < 8; ++t) {
        const int row0 = t + ((lane >> 4) << 3);
        const int row1 = row0 + 16;
        const float a0 = fx0[t] + bb, a1 = fx1[t] + bb;
        cf0[t] += sigm(a0 + p00[t]) * bf2f(c0T[row0 * 768 + col])
                + sigm(a0 + p10[t]) * bf2f(c1T[row0 * 768 + col]);
        cf1[t] += sigm(a1 + p01[t]) * bf2f(c0T[row1 * 768 + col])
                + sigm(a1 + p11[t]) * bf2f(c1T[row1 * 768 + col]);
      }
    }
    #pragma unroll
    for (int t = 0; t < 8; ++t) {
      const int row0 = t + ((lane >> 4) << 3);
      const int row1 = row0 + 16;
      cT[row0 * 768 + col] = f2bf(cf0[t]);
      oT[row0 * 768 + col] = f2bf(of0[t]);
      cT[row1 * 768 + col] = f2bf(cf1[t]);
      oT[row1 * 768 + col] = f2bf(of1[t]);
    }
  }
  __syncthreads();

  // ---- LayerNorm(c) -> h = sig(o)*tanh(.) -> LayerNorm(h); 8 lanes per row ----
  const int r = tid >> 3, s = tid & 7;          // 32 rows x 8 lanes
  const int base = r * 768 + s * 96;
  float sum = 0.f, sq = 0.f;
  for (int t = 0; t < 96; ++t) { const float v = bf2f(cT[base + t]); sum += v; sq += v * v; }
  for (int msk = 1; msk < 8; msk <<= 1) {
    sum += __shfl_xor(sum, msk, 32); sq += __shfl_xor(sq, msk, 32);
  }
  const float mu   = sum * (1.f / 768.f);
  const float rstd = rsqrtf(sq * (1.f / 768.f) - mu * mu + EPSF);
  float sum2 = 0.f, sq2 = 0.f;
  for (int t = 0; t < 96; ++t) {
    const int col = s * 96 + t;
    const float cn = (bf2f(cT[base + t]) - mu) * rstd * lncg[col] + lncb[col];
    const float hv = bf2f(oT[base + t]) * tanhf(cn);
    cT[base + t] = f2bf(hv);
    sum2 += hv; sq2 += hv * hv;
  }
  for (int msk = 1; msk < 8; msk <<= 1) {
    sum2 += __shfl_xor(sum2, msk, 32); sq2 += __shfl_xor(sq2, msk, 32);
  }
  const float mu2   = sum2 * (1.f / 768.f);
  const float rstd2 = rsqrtf(sq2 * (1.f / 768.f) - mu2 * mu2 + EPSF);
  unsigned short* op = hout + ((size_t)m * 32 + r) * 768 + s * 96;
  for (int t = 0; t < 96; ++t) {
    const int col = s * 96 + t;
    op[t] = f2bf((bf2f(cT[base + t]) - mu2) * rstd2 * lnhg[col] + lnhb[col]);
  }
}

// out = h_root @ Wout^T + b_out + x[:,0]   (128 x 768), one 16x16 tile per wave.
__global__ void __launch_bounds__(128)
treelstm_out(const unsigned short* __restrict__ hroot, const unsigned short* __restrict__ Wout,
             const float* __restrict__ bout,  const float* __restrict__ x,
             float* __restrict__ out) {
  const int wave = threadIdx.x >> 5, lane = threadIdx.x & 31;
  const int tileid = blockIdx.x * 4 + wave;   // 0..383
  const int mt = tileid & 7, nt = tileid >> 3;
  const int g = nt << 4;
  v8f acc = {};
  #pragma unroll 4
  for (int k0 = 0; k0 < 768; k0 += 32) {
    const v16bf a = a_frag_gbf(hroot, mt << 4, k0, lane);
    const v16bf b = b_frag(Wout, g, k0, lane);
    acc = WMMA_BF16(a, b, acc);
  }
  const int col = g + (lane & 15);
  const float bo = bout[col];
  #pragma unroll
  for (int t = 0; t < 8; ++t) {
    const int row = (mt << 4) + t + ((lane >> 4) << 3);
    out[(size_t)row * 768 + col] = acc[t] + bo + x[(size_t)row * NODES * 768 + col];
  }
}

static inline void launch_cvt(const float* s, unsigned short* d, long n, hipStream_t st) {
  const long n8 = n / 8;
  hipLaunchKernelGGL(cvt_bf16, dim3((unsigned)((n8 + 255) / 256)), dim3(256), 0, st, s, d, n8);
}

extern "C" void kernel_launch(void* const* d_in, const int* in_sizes, int n_in,
                              void* d_out, int out_size, void* d_ws, size_t ws_size,
                              hipStream_t stream) {
  const float* x     = (const float*)d_in[0];
  const float* Wioux = (const float*)d_in[1];
  const float* bioux = (const float*)d_in[2];
  const float* Wiouh = (const float*)d_in[3];
  const float* Wfx   = (const float*)d_in[4];
  const float* bfx   = (const float*)d_in[5];
  const float* Wfh   = (const float*)d_in[6];
  const float* bfh   = (const float*)d_in[7];
  const float* lncg  = (const float*)d_in[8];
  const float* lncb  = (const float*)d_in[9];
  const float* lnhg  = (const float*)d_in[10];
  const float* lnhb  = (const float*)d_in[11];
  const float* Wo    = (const float*)d_in[12];
  const float* bo    = (const float*)d_in[13];

  // ---- workspace carve (all bf16) ----
  const long NX   = (long)128 * NODES * 768;    // 25,067,520
  const long NWB  = (long)8 * 2304 * 768;       // 14,155,776
  const long NWS  = (long)8 * 768 * 768;        //  4,718,592
  const long NWO  = (long)768 * 768;            //    589,824
  const long NH   = (long)128 * 128 * 768;      // 12,582,912

  unsigned short* p = (unsigned short*)d_ws;
  unsigned short* xbf   = p; p += NX;
  unsigned short* wioux = p; p += NWB;
  unsigned short* wiouh = p; p += NWB;
  unsigned short* wfx   = p; p += NWS;
  unsigned short* wfh   = p; p += NWS;
  unsigned short* wout  = p; p += NWO;
  unsigned short* hA    = p; p += NH;
  unsigned short* hB    = p; p += NH;

  // ---- one-time fp32 -> bf16 conversion ----
  launch_cvt(x,     xbf,   NX,  stream);
  launch_cvt(Wioux, wioux, NWB, stream);
  launch_cvt(Wiouh, wiouh, NWB, stream);
  launch_cvt(Wfx,   wfx,   NWS, stream);
  launch_cvt(Wfh,   wfh,   NWS, stream);
  launch_cvt(Wo,    wout,  NWO, stream);

  const size_t smem = 6 * TILE2 * sizeof(unsigned short);  // 288 KB LDS

  unsigned short* hprev = nullptr;
  for (int l = 7; l >= 0; --l) {
    const int rows = 128 << l;
    unsigned short* ho = (l & 1) ? hA : hB;
    hipLaunchKernelGGL(treelstm_level, dim3(rows / 32), dim3(256), smem, stream,
        xbf,
        wioux + (size_t)l * 2304 * 768, bioux + (size_t)l * 2304,
        wiouh + (size_t)l * 2304 * 768,
        wfx   + (size_t)l * 768 * 768,  bfx   + (size_t)l * 768,
        wfh   + (size_t)l * 768 * 768,  bfh   + (size_t)l * 768,
        lncg  + (size_t)l * 768,        lncb  + (size_t)l * 768,
        lnhg  + (size_t)l * 768,        lnhb  + (size_t)l * 768,
        hprev, ho, l, (l < 7) ? 1 : 0);
    hprev = ho;
  }
  hipLaunchKernelGGL(treelstm_out, dim3(96), dim3(128), 0, stream,
                     hprev, wout, bo, x, (float*)d_out);
}
